// CausalMultiHeadSelfAttention_11553462026827
// MI455X (gfx1250) — compile-verified, matrix operand order checked by reading, not run on hardware
//
#include <hip/hip_runtime.h>
#include <stdint.h>

// Problem constants (match reference)
#define BATCHN 4
#define SEQL   2048
#define DMODEL 1024
#define HEADS  16
#define DKH    64
#define MTOK   (BATCHN * SEQL)   // 8192 tokens

typedef __bf16 bf16_t;
typedef bf16_t v16bf __attribute__((ext_vector_type(16)));
typedef bf16_t v8bf  __attribute__((ext_vector_type(8)));
typedef float  v8f   __attribute__((ext_vector_type(8)));

__device__ __forceinline__ bf16_t f2bf(float f) {
  union { float f; uint32_t u; } c; c.f = f;
  uint32_t r = (c.u + 0x7FFFu + ((c.u >> 16) & 1u)) >> 16;  // RNE
  union { uint16_t s; bf16_t b; } o; o.s = (uint16_t)r;
  return o.b;
}

// Load a 16x(K=32) A/B fragment half for this lane: 8 contiguous bf16 at p,
// 8 contiguous bf16 at p+16 (ISA 16-bit A/B layout: K = base+0..7, base+16..23).
__device__ __forceinline__ v16bf load_frag(const bf16_t* p) {
  v8bf lo = *(const v8bf*)(p);
  v8bf hi = *(const v8bf*)(p + 16);
  v16bf r;
#pragma unroll
  for (int i = 0; i < 8; i++) { r[i] = lo[i]; r[i + 8] = hi[i]; }
  return r;
}

__device__ __forceinline__ float redmax16(float x) {
#pragma unroll
  for (int off = 1; off < 16; off <<= 1) x = fmaxf(x, __shfl_xor(x, off, 32));
  return x;
}
__device__ __forceinline__ float redsum16(float x) {
#pragma unroll
  for (int off = 1; off < 16; off <<= 1) x += __shfl_xor(x, off, 32);
  return x;
}

// ---------------- conversion kernels ----------------

__global__ __launch_bounds__(256) void cvt_f32_to_bf16_k(const float* __restrict__ in,
                                                         bf16_t* __restrict__ out, int n) {
  int i = blockIdx.x * blockDim.x + threadIdx.x;
  if (i < n) out[i] = f2bf(in[i]);
}

// in: [K][N] fp32 (row-major) -> out: [N][K] bf16 (transposed)
__global__ __launch_bounds__(256) void cvt_transpose_k(const float* __restrict__ in,
                                                       bf16_t* __restrict__ out, int K, int N) {
  int i = blockIdx.x * blockDim.x + threadIdx.x;
  if (i >= K * N) return;
  int k = i / N;
  int n = i - k * N;               // n fastest -> coalesced read
  out[(size_t)n * K + k] = f2bf(in[(size_t)k * N + n]);
}

// ---------------- WMMA GEMM: C[MxN] = (A[MxK](bf16) * BT[NxK]^T + bias) * outScale ----
// Wave tile 32M x 64N (2 A-frags x 4 B-frags -> 8 wmma / K-step); loads hoisted
// ahead of the wmma group for clause overlap. 8 waves as 4(M) x 2(N) -> 128x128.
// MODE 0: store bf16 [MxN]
// MODE 1: store bf16 transposed per-head Vt[B,H,DKH,SEQL]  (for attention PV fragments)
// MODE 2: store fp32 [MxN]
template <int MODE>
__global__ __launch_bounds__(256) void gemm_bf16_wmma_k(const bf16_t* __restrict__ A,
                                                        const bf16_t* __restrict__ BT,
                                                        const float* __restrict__ bias,
                                                        void* __restrict__ Cout,
                                                        int M, int N, int K,
                                                        float outScale) {
  const int lane = threadIdx.x & 31;
  const int wave = threadIdx.x >> 5;
  const int waveM = wave & 3;
  const int waveN = wave >> 2;
  const int lc   = lane & 15;
  const int half = lane >> 4;              // K-offset selector per ISA layout
  const int mBlock = blockIdx.x * 128 + waveM * 32;
  const int nBlock = blockIdx.y * 128 + waveN * 64;

  v8f acc[2][4];
#pragma unroll
  for (int r = 0; r < 2; r++)
#pragma unroll
    for (int t = 0; t < 4; t++) acc[r][t] = v8f{};

  const bf16_t* Arow0 = A + (size_t)(mBlock + lc) * K + half * 8;
  const bf16_t* Arow1 = A + (size_t)(mBlock + 16 + lc) * K + half * 8;
  const bf16_t* Brow0 = BT + (size_t)(nBlock + lc) * K + half * 8;

  for (int k0 = 0; k0 < K; k0 += 32) {
    // hoisted load clause: 12 x global_load_b128 into distinct registers
    v16bf a0 = load_frag(Arow0 + k0);
    v16bf a1 = load_frag(Arow1 + k0);
    v16bf b[4];
#pragma unroll
    for (int t = 0; t < 4; t++) b[t] = load_frag(Brow0 + (size_t)(t * 16) * K + k0);
    // 8 back-to-back wmma
#pragma unroll
    for (int t = 0; t < 4; t++) {
      acc[0][t] = __builtin_amdgcn_wmma_f32_16x16x32_bf16(
          false, a0, false, b[t], (short)0, acc[0][t], false, false);
      acc[1][t] = __builtin_amdgcn_wmma_f32_16x16x32_bf16(
          false, a1, false, b[t], (short)0, acc[1][t], false, false);
    }
  }

#pragma unroll
  for (int t = 0; t < 4; t++) {
    const int ncol = nBlock + t * 16 + lc;
    const float bv = bias[ncol];
#pragma unroll
    for (int r = 0; r < 2; r++) {
#pragma unroll
      for (int v = 0; v < 8; v++) {
        const int m = mBlock + r * 16 + v + 8 * half;  // C layout row mapping
        const float val = (acc[r][t][v] + bv) * outScale;
        if (MODE == 2) {
          ((float*)Cout)[(size_t)m * N + ncol] = val;
        } else if (MODE == 0) {
          ((bf16_t*)Cout)[(size_t)m * N + ncol] = f2bf(val);
        } else {
          // Vt[B,H,DKH,SEQL]: ncol -> (h,d); m -> (b,s)
          const int h = ncol >> 6, d = ncol & 63;
          const int bb = m >> 11, s = m & 2047;   // SEQL = 2048
          ((bf16_t*)Cout)[(((size_t)bb * HEADS + h) * DKH + d) * SEQL + s] = f2bf(val);
        }
      }
    }
  }
}

// ---------------- Flash attention (causal, online softmax, all WMMA) ----------------
// One 64-key block step. Q is pre-scaled by 1/sqrt(d_k) in its projection, so no
// per-score multiply here. MASKED=true only for the diagonal block.
// The P C-layout -> A-layout transpose uses a wave-private LDS tile: DS ops are
// in-order per wave, so only a compiler fence (wave_barrier) is needed.
template <bool MASKED>
__device__ __forceinline__ void attn_step(int j0, int qbase, int lc, int half,
                                          const bf16_t* __restrict__ Kbase,
                                          const bf16_t* __restrict__ Vbase,
                                          const v16bf& q0, const v16bf& q1,
                                          bf16_t (*__restrict__ pw)[64],  // [16][64] tile
                                          v8f o[4], float mrow[8], float lrow[8]) {
  // ---- load all 8 K fragments as one clause ----
  v16bf kb[8];
#pragma unroll
  for (int kt = 0; kt < 4; kt++) {
    const bf16_t* Kp = Kbase + (size_t)(j0 + kt * 16) * DMODEL;
    kb[2 * kt]     = load_frag(Kp);
    kb[2 * kt + 1] = load_frag(Kp + 32);
  }

  // ---- scores: S = Q(16x64) * K^T(64x64) as four 16x16 tiles, 8 wmma ----
  v8f s[4];
#pragma unroll
  for (int kt = 0; kt < 4; kt++) {
    v8f sv = __builtin_amdgcn_wmma_f32_16x16x32_bf16(false, q0, false, kb[2 * kt],
                                                     (short)0, v8f{}, false, false);
    s[kt] = __builtin_amdgcn_wmma_f32_16x16x32_bf16(false, q1, false, kb[2 * kt + 1],
                                                    (short)0, sv, false, false);
  }

  // ---- issue V fragment loads now: in flight during softmax + LDS transpose ----
  v16bf vb[8];
#pragma unroll
  for (int t = 0; t < 4; t++) {
    const bf16_t* Vp = Vbase + (size_t)(t * 16) * SEQL + j0;
    vb[2 * t]     = load_frag(Vp);
    vb[2 * t + 1] = load_frag(Vp + 32);
  }

  // ---- (masked) online softmax: one butterfly pass covers 64 keys ----
  float p[4][8];
#pragma unroll
  for (int v = 0; v < 8; v++) {
    float a[4];
#pragma unroll
    for (int kt = 0; kt < 4; kt++) {
      if (MASKED) {
        const int qr = qbase + v + 8 * half;
        const int ki = j0 + kt * 16 + lc;
        a[kt] = (ki <= qr) ? s[kt][v] : -1.0e30f;
      } else {
        a[kt] = s[kt][v];
      }
    }
    const float mx = redmax16(fmaxf(fmaxf(a[0], a[1]), fmaxf(a[2], a[3])));
    const float mnew = fmaxf(mrow[v], mx);
    float e[4], esum = 0.f;
#pragma unroll
    for (int kt = 0; kt < 4; kt++) { e[kt] = __expf(a[kt] - mnew); esum += e[kt]; }
    const float rs = redsum16(esum);
    const float corr = __expf(mrow[v] - mnew);
    lrow[v] = lrow[v] * corr + rs;
    mrow[v] = mnew;
#pragma unroll
    for (int kt = 0; kt < 4; kt++) p[kt][v] = e[kt];
#pragma unroll
    for (int t = 0; t < 4; t++) o[t][v] *= corr;
  }

  // ---- transpose P through wave-private LDS tile ----
#pragma unroll
  for (int v = 0; v < 8; v++) {
    const int r = v + 8 * half;
#pragma unroll
    for (int kt = 0; kt < 4; kt++) pw[r][kt * 16 + lc] = f2bf(p[kt][v]);
  }
  __builtin_amdgcn_wave_barrier();   // code-motion fence; DS is in-order per wave
  const v16bf pa0 = load_frag(&pw[lc][half * 8]);        // keys j0+0..31
  const v16bf pa1 = load_frag(&pw[lc][half * 8 + 32]);   // keys j0+32..63
  __builtin_amdgcn_wave_barrier();

  // ---- O += P(16x64) * V(64x64): 8 back-to-back wmma ----
#pragma unroll
  for (int t = 0; t < 4; t++) {
    o[t] = __builtin_amdgcn_wmma_f32_16x16x32_bf16(false, pa0, false, vb[2 * t],
                                                   (short)0, o[t], false, false);
    o[t] = __builtin_amdgcn_wmma_f32_16x16x32_bf16(false, pa1, false, vb[2 * t + 1],
                                                   (short)0, o[t], false, false);
  }
}

// grid: (SEQL/64, BATCHN*HEADS), block: 128 threads = 4 waves; wave owns 16 query rows.
__global__ __launch_bounds__(128) void flash_attn_wmma_k(const bf16_t* __restrict__ Qg,
                                                         const bf16_t* __restrict__ Kg,
                                                         const bf16_t* __restrict__ Vt,
                                                         bf16_t* __restrict__ Yg) {
  __shared__ bf16_t plds[4][16][64];  // per-wave P tiles

  const int lane = threadIdx.x & 31;
  const int wave = threadIdx.x >> 5;
  const int lc   = lane & 15;
  const int half = lane >> 4;
  const int bh = blockIdx.y;
  const int b  = bh >> 4;
  const int h  = bh & 15;
  const int qtile = blockIdx.x;            // 64 queries per block
  const int qbase = qtile * 64 + wave * 16;

  // Q A-fragments (already scaled by 1/sqrt(d_k) at projection time)
  const bf16_t* Qp = Qg + ((size_t)(b * SEQL + qbase + lc)) * DMODEL + h * DKH + half * 8;
  const v16bf q0 = load_frag(Qp);        // d 0..31
  const v16bf q1 = load_frag(Qp + 32);   // d 32..63

  const bf16_t* Kbase = Kg + ((size_t)(b * SEQL + lc)) * DMODEL + h * DKH + half * 8;
  const bf16_t* Vbase = Vt + (((size_t)(b * HEADS + h)) * DKH + lc) * SEQL + half * 8;

  v8f o[4] = {v8f{}, v8f{}, v8f{}, v8f{}};
  float mrow[8], lrow[8];
#pragma unroll
  for (int v = 0; v < 8; v++) { mrow[v] = -1.0e30f; lrow[v] = 0.0f; }

  // below-diagonal blocks: all 64 keys valid for every query in this block
  const int jdiag = qtile * 64;
  for (int j0 = 0; j0 < jdiag; j0 += 64)
    attn_step<false>(j0, qbase, lc, half, Kbase, Vbase, q0, q1, plds[wave], o, mrow, lrow);
  // diagonal block: causal masking required
  attn_step<true>(jdiag, qbase, lc, half, Kbase, Vbase, q0, q1, plds[wave], o, mrow, lrow);

  // ---- normalize and emit y[B,S,D] bf16 ----
#pragma unroll
  for (int t = 0; t < 4; t++) {
#pragma unroll
    for (int v = 0; v < 8; v++) {
      const int qr = qbase + v + 8 * half;
      const float val = o[t][v] / lrow[v];
      Yg[((size_t)(b * SEQL + qr)) * DMODEL + h * DKH + t * 16 + lc] = f2bf(val);
    }
  }
}

// ---------------- host-side orchestration ----------------

extern "C" void kernel_launch(void* const* d_in, const int* in_sizes, int n_in,
                              void* d_out, int out_size, void* d_ws, size_t ws_size,
                              hipStream_t stream) {
  (void)in_sizes; (void)n_in; (void)out_size; (void)ws_size;

  const float* x   = (const float*)d_in[0];
  const float* w_q = (const float*)d_in[1];
  const float* b_q = (const float*)d_in[2];
  const float* w_k = (const float*)d_in[3];
  const float* b_k = (const float*)d_in[4];
  const float* w_v = (const float*)d_in[5];
  const float* b_v = (const float*)d_in[6];
  const float* w_o = (const float*)d_in[7];
  const float* b_o = (const float*)d_in[8];
  float* out = (float*)d_out;

  const size_t MB = 1024ull * 1024ull;
  char* ws = (char*)d_ws;
  bf16_t* xb  = (bf16_t*)(ws + 0);          // 16 MB  : x in bf16 [MTOK x D]
  bf16_t* wqT = (bf16_t*)(ws + 16 * MB);    // 2 MB   : w_q^T bf16 [N x K]
  bf16_t* wkT = (bf16_t*)(ws + 18 * MB);
  bf16_t* wvT = (bf16_t*)(ws + 20 * MB);
  bf16_t* woT = (bf16_t*)(ws + 22 * MB);
  bf16_t* Qb  = (bf16_t*)(ws + 24 * MB);    // 16 MB  : Q bf16 [MTOK x D] (pre-scaled)
  bf16_t* Kb  = (bf16_t*)(ws + 40 * MB);    // 16 MB  : K bf16 [MTOK x D]
  bf16_t* Vtb = (bf16_t*)(ws + 56 * MB);    // 16 MB  : V^T bf16 [B,H,DKH,SEQL]
  bf16_t* Yb  = (bf16_t*)(ws + 72 * MB);    // 16 MB  : attention output bf16 [MTOK x D]

  const int M = MTOK, N = DMODEL, K = DMODEL;

  // 1) convert activations + weights to bf16 (weights transposed for B-fragment reads)
  {
    const int n = M * K;
    cvt_f32_to_bf16_k<<<(n + 255) / 256, 256, 0, stream>>>(x, xb, n);
    const int wn = K * N;
    const int wg = (wn + 255) / 256;
    cvt_transpose_k<<<wg, 256, 0, stream>>>(w_q, wqT, K, N);
    cvt_transpose_k<<<wg, 256, 0, stream>>>(w_k, wkT, K, N);
    cvt_transpose_k<<<wg, 256, 0, stream>>>(w_v, wvT, K, N);
    cvt_transpose_k<<<wg, 256, 0, stream>>>(w_o, woT, K, N);
  }

  // 2) QKV projections. Q pre-scaled by 1/sqrt(d_k); V stored transposed per head.
  {
    dim3 grid(M / 128, N / 128);
    gemm_bf16_wmma_k<0><<<grid, 256, 0, stream>>>(xb, wqT, b_q, Qb,  M, N, K, 0.125f);
    gemm_bf16_wmma_k<0><<<grid, 256, 0, stream>>>(xb, wkT, b_k, Kb,  M, N, K, 1.0f);
    gemm_bf16_wmma_k<1><<<grid, 256, 0, stream>>>(xb, wvT, b_v, Vtb, M, N, K, 1.0f);
  }

  // 3) causal flash attention (all WMMA)
  {
    dim3 grid(SEQL / 64, BATCHN * HEADS);
    flash_attn_wmma_k<<<grid, 128, 0, stream>>>(Qb, Kb, Vtb, Yb);
  }

  // 4) output projection -> fp32 out
  {
    dim3 grid(M / 128, N / 128);
    gemm_bf16_wmma_k<2><<<grid, 256, 0, stream>>>(Yb, woT, b_o, out, M, N, K, 1.0f);
  }
}
